// PolyINRWithHypernet_77850577207560
// MI455X (gfx1250) — matrix operand order repaired and checked by reading, not compiled, not run on hardware
//
#include <hip/hip_runtime.h>
#include <hip/hip_bf16.h>
#include <math.h>
#include <stdint.h>

#define B_SZ 32768
#define HIN  128
#define HH   512
#define DH   256
#define NL   5

#define BM 128
#define BN 128
#define BK 32

typedef __attribute__((ext_vector_type(16))) __bf16 v16bf;
typedef __attribute__((ext_vector_type(8)))  float  v8f;

__device__ __forceinline__ __bf16 f2bf(float x) { return (__bf16)x; }
__device__ __forceinline__ float  bf2f(__bf16 x) { return (float)x; }
__device__ __forceinline__ float gelu_exact(float x) {
    return 0.5f * x * (1.0f + erff(x * 0.70710678118654752f));
}

// ---- CDNA5 async global->LDS copy (ASYNCcnt-tracked), per-lane 16B
__device__ __forceinline__ void async_ld_b128(uint32_t lds_off, const void* gptr) {
    asm volatile("global_load_async_to_lds_b128 %0, %1, off"
                 :: "v"(lds_off), "v"((unsigned long long)(uintptr_t)gptr)
                 : "memory");
}
template<int N>
__device__ __forceinline__ void wait_async() {
    asm volatile("s_wait_asynccnt %0" :: "i"(N) : "memory");
}

union FragU { v16bf v; uint4 q[2]; };
union FragC { v8f v; float f[8]; };
union Pack8 { uint4 q; __bf16 b[8]; };

// A-operand fragment (16 rows x 32 K). lane r(<16): M=r, K={0..7,16..23};
// lane r+16: M=r, K={8..15,24..31}.
__device__ __forceinline__ v16bf load_frag_a(const __bf16* tile, int lane) {
    int r = lane & 15, half = lane >> 4;
    const __bf16* p = tile + r * BK + half * 8;
    FragU f;
    f.q[0] = *(const uint4*)(p);
    f.q[1] = *(const uint4*)(p + 16);
    return f.v;
}
// B-operand fragment from [N][K] tile. lane r(<16): N=r, K={0..7,16..23}; lane r+16: K={8..15,24..31}.
__device__ __forceinline__ v16bf load_frag_b(const __bf16* tile, int lane) {
    int r = lane & 15, half = lane >> 4;
    const __bf16* p = tile + r * BK + half * 8;
    FragU f;
    f.q[0] = *(const uint4*)(p);
    f.q[1] = *(const uint4*)(p + 16);
    return f.v;
}

// ---------------------------------------------------------------- prep kernels
__global__ __launch_bounds__(256)
void cvt_kernel(const float* __restrict__ src, __bf16* __restrict__ dst, int n) {
    int i = blockIdx.x * 256 + threadIdx.x;
    if (i < n) dst[i] = f2bf(src[i]);
}

// src: [NL][R][C] f32 -> dst: [NL][C][R] bf16
__global__ void cvt_transpose_kernel(const float* __restrict__ src,
                                     __bf16* __restrict__ dst, int R, int C) {
    int l = blockIdx.z;
    size_t off = (size_t)l * R * C;
    int c = blockIdx.x * 32 + threadIdx.x;
    int r = blockIdx.y * 8 + threadIdx.y;
    if (c < C && r < R)
        dst[off + (size_t)c * R + r] = f2bf(src[off + (size_t)r * C + c]);
}

// ------------------------------------------------------------- generic GEMM
// C = act(A[M,K] * BT[N,K]^T + bias[N]);  EPI=0: gelu -> bf16, EPI=1: none -> f32
template<int EPI>
__global__ __launch_bounds__(256)
void gemm_bf16_kernel(const __bf16* __restrict__ A, const __bf16* __restrict__ BT,
                      const float* __restrict__ bias, void* __restrict__ outp,
                      int M, int N, int K) {
    __shared__ __align__(16) __bf16 sA[2][BM * BK];
    __shared__ __align__(16) __bf16 sB[2][BN * BK];
    const int tid  = threadIdx.x;
    const int lane = tid & 31;
    const int wave = tid >> 5;
    const int wm = wave >> 1;          // 0..3 (M, 32 rows each)
    const int wn = wave & 1;           // 0..1 (N, 64 cols each)
    const int bm0 = blockIdx.y * BM;
    const int bn0 = blockIdx.x * BN;

    FragC acc[2][4];
#pragma unroll
    for (int i = 0; i < 2; ++i)
#pragma unroll
        for (int j = 0; j < 4; ++j) acc[i][j].v = (v8f){0,0,0,0,0,0,0,0};

    const int lr = tid >> 2;           // 0..63
    const int lc = (tid & 3) * 8;      // 0,8,16,24

    const __bf16* gA = A  + (size_t)(bm0 + lr) * K + lc;
    const __bf16* gB = BT + (size_t)(bn0 + lr) * K + lc;
    const size_t rowskip = (size_t)64 * K;
    const uint32_t lA0 = (uint32_t)(uintptr_t)&sA[0][lr * BK + lc];
    const uint32_t lA1 = (uint32_t)(uintptr_t)&sA[0][(lr + 64) * BK + lc];
    const uint32_t lB0 = (uint32_t)(uintptr_t)&sB[0][lr * BK + lc];
    const uint32_t lB1 = (uint32_t)(uintptr_t)&sB[0][(lr + 64) * BK + lc];
    const uint32_t bufstep = (uint32_t)(BM * BK * 2);   // bytes between buffers

    const int nk = K / BK;

    // prologue: async-load tile 0 into buffer 0
    {
        async_ld_b128(lA0, gA);
        async_ld_b128(lA1, gA + rowskip);
        async_ld_b128(lB0, gB);
        async_ld_b128(lB1, gB + rowskip);
    }

    for (int kt = 0; kt < nk; ++kt) {
        const int cur = kt & 1;
        if (kt + 1 < nk) {
            // issue next tile into the other buffer, then wait for current tile
            const __bf16* a = gA + (size_t)(kt + 1) * BK;
            const __bf16* b = gB + (size_t)(kt + 1) * BK;
            const uint32_t o = (uint32_t)(cur ^ 1) * bufstep;
            async_ld_b128(lA0 + o, a);
            async_ld_b128(lA1 + o, a + rowskip);
            async_ld_b128(lB0 + o, b);
            async_ld_b128(lB1 + o, b + rowskip);
            wait_async<4>();           // 4 newest in flight -> current tile done
        } else {
            wait_async<0>();
        }
        __syncthreads();

        v16bf af[2], bfv[4];
#pragma unroll
        for (int i = 0; i < 2; ++i)
            af[i] = load_frag_a(&sA[cur][(wm * 32 + i * 16) * BK], lane);
#pragma unroll
        for (int j = 0; j < 4; ++j)
            bfv[j] = load_frag_b(&sB[cur][(wn * 64 + j * 16) * BK], lane);
#pragma unroll
        for (int i = 0; i < 2; ++i)
#pragma unroll
            for (int j = 0; j < 4; ++j)
                acc[i][j].v = __builtin_amdgcn_wmma_f32_16x16x32_bf16(
                    false, af[i], false, bfv[j], (short)0, acc[i][j].v, false, false);
        __syncthreads();
    }

#pragma unroll
    for (int i = 0; i < 2; ++i)
#pragma unroll
        for (int j = 0; j < 4; ++j) {
            int col   = bn0 + wn * 64 + j * 16 + (lane & 15);
            int rbase = bm0 + wm * 32 + i * 16 + (lane >> 4) * 8;
            float bv = bias[col];
#pragma unroll
            for (int v = 0; v < 8; ++v) {
                int row = rbase + v;
                float h = acc[i][j].f[v] + bv;
                if (EPI == 0)
                    ((__bf16*)outp)[(size_t)row * N + col] = f2bf(gelu_exact(h));
                else
                    ((float*)outp)[(size_t)row * N + col] = h;
            }
        }
}

// ------------------------------------------------------------- INR trunk layer
// hidden' = gelu((1+scale)*((hidden .* (x@affW+affB)) @ denseW + denseB) + shift)
template<int HASHID>
__global__ __launch_bounds__(256)
void inr_layer_kernel(const __bf16* __restrict__ hidIn,
                      const float* __restrict__ coord,
                      const float* __restrict__ affW, const float* __restrict__ affB,
                      const __bf16* __restrict__ denseT, const float* __restrict__ denseB,
                      const float* __restrict__ scaleM, const float* __restrict__ shiftM,
                      __bf16* __restrict__ hidOut) {
    __shared__ __align__(16) __bf16 sA[2][BM * BK];
    __shared__ __align__(16) __bf16 sB[2][BN * BK];
    const int tid  = threadIdx.x;
    const int lane = tid & 31;
    const int wave = tid >> 5;
    const int wm = wave >> 1;
    const int wn = wave & 1;
    const int bm0 = blockIdx.y * BM;
    const int bn0 = blockIdx.x * BN;

    FragC acc[2][4];
#pragma unroll
    for (int i = 0; i < 2; ++i)
#pragma unroll
        for (int j = 0; j < 4; ++j) acc[i][j].v = (v8f){0,0,0,0,0,0,0,0};

    const int lr = tid >> 2;
    const int lc = (tid & 3) * 8;

    const __bf16* gB = denseT + (size_t)(bn0 + lr) * DH + lc;
    const size_t rowskip = (size_t)64 * DH;
    const uint32_t lB0 = (uint32_t)(uintptr_t)&sB[0][lr * BK + lc];
    const uint32_t lB1 = (uint32_t)(uintptr_t)&sB[0][(lr + 64) * BK + lc];
    const uint32_t bufstep = (uint32_t)(BN * BK * 2);

    // per-row constants for the A-tile builder (rows lr and lr+64)
    float x0[2], x1[2];
    const __bf16* hrow[2];
#pragma unroll
    for (int rr = 0; rr < 2; ++rr) {
        int row = bm0 + lr + rr * 64;
        x0[rr] = coord[(size_t)row * 2 + 0];
        x1[rr] = coord[(size_t)row * 2 + 1];
        hrow[rr] = HASHID ? (hidIn + (size_t)row * DH) : (const __bf16*)nullptr;
    }

    const int nk = DH / BK;
    // prologue: B tile 0
    async_ld_b128(lB0, gB);
    async_ld_b128(lB1, gB + rowskip);

    for (int kt = 0; kt < nk; ++kt) {
        const int cur = kt & 1;
        const int kbase = kt * BK + lc;

        // build modulated A tile (vectorized, branch-free)
        float4 w0a = *(const float4*)(affW + kbase);
        float4 w0b = *(const float4*)(affW + kbase + 4);
        float4 w1a = *(const float4*)(affW + DH + kbase);
        float4 w1b = *(const float4*)(affW + DH + kbase + 4);
        float4 bba = *(const float4*)(affB + kbase);
        float4 bbb = *(const float4*)(affB + kbase + 4);
        float w0[8] = {w0a.x, w0a.y, w0a.z, w0a.w, w0b.x, w0b.y, w0b.z, w0b.w};
        float w1[8] = {w1a.x, w1a.y, w1a.z, w1a.w, w1b.x, w1b.y, w1b.z, w1b.w};
        float bb[8] = {bba.x, bba.y, bba.z, bba.w, bbb.x, bbb.y, bbb.z, bbb.w};
#pragma unroll
        for (int rr = 0; rr < 2; ++rr) {
            Pack8 hv, o;
            if (HASHID) hv.q = *(const uint4*)(hrow[rr] + kbase);
#pragma unroll
            for (int e = 0; e < 8; ++e) {
                float t = x0[rr] * w0[e] + x1[rr] * w1[e] + bb[e];
                float h = HASHID ? (bf2f(hv.b[e]) * t) : t;
                o.b[e] = f2bf(h);
            }
            *(uint4*)&sA[cur][(lr + rr * 64) * BK + lc] = o.q;
        }

        if (kt + 1 < nk) {
            const __bf16* b = gB + (size_t)(kt + 1) * BK;
            const uint32_t off = (uint32_t)(cur ^ 1) * bufstep;
            async_ld_b128(lB0 + off, b);
            async_ld_b128(lB1 + off, b + rowskip);
            wait_async<2>();
        } else {
            wait_async<0>();
        }
        __syncthreads();

        v16bf af[2], bfv[4];
#pragma unroll
        for (int i = 0; i < 2; ++i)
            af[i] = load_frag_a(&sA[cur][(wm * 32 + i * 16) * BK], lane);
#pragma unroll
        for (int j = 0; j < 4; ++j)
            bfv[j] = load_frag_b(&sB[cur][(wn * 64 + j * 16) * BK], lane);
#pragma unroll
        for (int i = 0; i < 2; ++i)
#pragma unroll
            for (int j = 0; j < 4; ++j)
                acc[i][j].v = __builtin_amdgcn_wmma_f32_16x16x32_bf16(
                    false, af[i], false, bfv[j], (short)0, acc[i][j].v, false, false);
        __syncthreads();
    }

#pragma unroll
    for (int i = 0; i < 2; ++i)
#pragma unroll
        for (int j = 0; j < 4; ++j) {
            int col   = bn0 + wn * 64 + j * 16 + (lane & 15);
            int rbase = bm0 + wm * 32 + i * 16 + (lane >> 4) * 8;
            float bv = denseB[col];
#pragma unroll
            for (int v = 0; v < 8; ++v) {
                int row = rbase + v;
                size_t mi = (size_t)row * DH + col;
                float h = acc[i][j].f[v] + bv;
                h = (1.0f + scaleM[mi]) * h + shiftM[mi];
                hidOut[mi] = f2bf(gelu_exact(h));
            }
        }
}

// ------------------------------------------------------------- final projection
__global__ __launch_bounds__(256)
void inr_last_kernel(const __bf16* __restrict__ hid, const float* __restrict__ W,
                     const float* __restrict__ b, float* __restrict__ out) {
    int lane = threadIdx.x & 31;
    int wave = threadIdx.x >> 5;
    int row = blockIdx.x * 8 + wave;
    const __bf16* hrow = hid + (size_t)row * DH;
    float s = 0.0f;
#pragma unroll
    for (int k = lane; k < DH; k += 32) s += bf2f(hrow[k]) * W[k];
#pragma unroll
    for (int off = 16; off; off >>= 1) s += __shfl_down(s, off, 32);
    if (lane == 0) out[row] = s + b[0];
}

// ----------------------------------------------------------------- host driver
extern "C" void kernel_launch(void* const* d_in, const int* in_sizes, int n_in,
                              void* d_out, int out_size, void* d_ws, size_t ws_size,
                              hipStream_t stream) {
    (void)in_sizes; (void)n_in; (void)out_size; (void)ws_size;
    const float* hyper_in = (const float*)d_in[0];
    const float* coord    = (const float*)d_in[1];
    const float* affW     = (const float*)d_in[2];   // [NL,2,DH]
    const float* affB     = (const float*)d_in[3];   // [NL,DH]
    const float* denseW   = (const float*)d_in[4];   // [NL,DH,DH]
    const float* denseB   = (const float*)d_in[5];   // [NL,DH]
    const float* lastW    = (const float*)d_in[6];   // [DH,1]
    const float* lastB    = (const float*)d_in[7];   // [1]
    const float* sW1 = (const float*)d_in[8],  *sB1 = (const float*)d_in[9];
    const float* sW2 = (const float*)d_in[10], *sB2 = (const float*)d_in[11];
    const float* sW3 = (const float*)d_in[12], *sB3 = (const float*)d_in[13];
    const float* cW1 = (const float*)d_in[14], *cB1 = (const float*)d_in[15];
    const float* cW2 = (const float*)d_in[16], *cB2 = (const float*)d_in[17];
    const float* cW3 = (const float*)d_in[18], *cB3 = (const float*)d_in[19];

    char* ws = (char*)d_ws;
    size_t off = 0;
    auto take = [&](size_t bytes) -> void* {
        void* p = ws + off;
        off += (bytes + 255) & ~(size_t)255;
        return p;
    };
    __bf16* hyperB   = (__bf16*)take((size_t)B_SZ * HIN * 2);
    __bf16* H1       = (__bf16*)take((size_t)B_SZ * HH * 2);
    __bf16* H2       = (__bf16*)take((size_t)B_SZ * HH * 2);
    float*  modShift = (float*) take((size_t)B_SZ * DH * 4);
    float*  modScale = (float*) take((size_t)B_SZ * DH * 4);
    __bf16* hidA     = (__bf16*)take((size_t)B_SZ * DH * 2);
    __bf16* hidB     = (__bf16*)take((size_t)B_SZ * DH * 2);
    __bf16* w1Ts     = (__bf16*)take((size_t)NL * HIN * HH * 2);
    __bf16* w1Tc     = (__bf16*)take((size_t)NL * HIN * HH * 2);
    __bf16* w2Ts     = (__bf16*)take((size_t)NL * HH * HH * 2);
    __bf16* w2Tc     = (__bf16*)take((size_t)NL * HH * HH * 2);
    __bf16* w3Ts     = (__bf16*)take((size_t)NL * HH * DH * 2);
    __bf16* w3Tc     = (__bf16*)take((size_t)NL * HH * DH * 2);
    __bf16* dT       = (__bf16*)take((size_t)NL * DH * DH * 2);

    cvt_kernel<<<(B_SZ * HIN) / 256, 256, 0, stream>>>(hyper_in, hyperB, B_SZ * HIN);
    dim3 tb(32, 8);
    cvt_transpose_kernel<<<dim3(HH / 32, HIN / 8, NL), tb, 0, stream>>>(sW1, w1Ts, HIN, HH);
    cvt_transpose_kernel<<<dim3(HH / 32, HIN / 8, NL), tb, 0, stream>>>(cW1, w1Tc, HIN, HH);
    cvt_transpose_kernel<<<dim3(HH / 32, HH / 8, NL),  tb, 0, stream>>>(sW2, w2Ts, HH, HH);
    cvt_transpose_kernel<<<dim3(HH / 32, HH / 8, NL),  tb, 0, stream>>>(cW2, w2Tc, HH, HH);
    cvt_transpose_kernel<<<dim3(DH / 32, HH / 8, NL),  tb, 0, stream>>>(sW3, w3Ts, HH, DH);
    cvt_transpose_kernel<<<dim3(DH / 32, HH / 8, NL),  tb, 0, stream>>>(cW3, w3Tc, HH, DH);
    cvt_transpose_kernel<<<dim3(DH / 32, DH / 8, NL),  tb, 0, stream>>>(denseW, dT, DH, DH);

    const dim3 gL1(HH / BN, B_SZ / BM), gL3(DH / BN, B_SZ / BM);
    for (int l = 0; l < NL; ++l) {
        // shift hypernet: 128 -> 512 -> 512 -> 256
        gemm_bf16_kernel<0><<<gL1, 256, 0, stream>>>(hyperB, w1Ts + (size_t)l * HH * HIN,
                                                     sB1 + (size_t)l * HH, (void*)H1, B_SZ, HH, HIN);
        gemm_bf16_kernel<0><<<gL1, 256, 0, stream>>>(H1, w2Ts + (size_t)l * HH * HH,
                                                     sB2 + (size_t)l * HH, (void*)H2, B_SZ, HH, HH);
        gemm_bf16_kernel<1><<<gL3, 256, 0, stream>>>(H2, w3Ts + (size_t)l * DH * HH,
                                                     sB3 + (size_t)l * DH, (void*)modShift, B_SZ, DH, HH);
        // scale hypernet
        gemm_bf16_kernel<0><<<gL1, 256, 0, stream>>>(hyperB, w1Tc + (size_t)l * HH * HIN,
                                                     cB1 + (size_t)l * HH, (void*)H1, B_SZ, HH, HIN);
        gemm_bf16_kernel<0><<<gL1, 256, 0, stream>>>(H1, w2Tc + (size_t)l * HH * HH,
                                                     cB2 + (size_t)l * HH, (void*)H2, B_SZ, HH, HH);
        gemm_bf16_kernel<1><<<gL3, 256, 0, stream>>>(H2, w3Tc + (size_t)l * DH * HH,
                                                     cB3 + (size_t)l * DH, (void*)modScale, B_SZ, DH, HH);
        // INR trunk layer l
        __bf16* hout = (l & 1) ? hidB : hidA;
        if (l == 0) {
            inr_layer_kernel<0><<<gL3, 256, 0, stream>>>(
                (const __bf16*)nullptr, coord,
                affW + (size_t)l * 2 * DH, affB + (size_t)l * DH,
                dT + (size_t)l * DH * DH, denseB + (size_t)l * DH,
                modScale, modShift, hout);
        } else {
            const __bf16* hin = (l & 1) ? hidA : hidB;
            inr_layer_kernel<1><<<gL3, 256, 0, stream>>>(
                hin, coord,
                affW + (size_t)l * 2 * DH, affB + (size_t)l * DH,
                dT + (size_t)l * DH * DH, denseB + (size_t)l * DH,
                modScale, modShift, hout);
        }
    }
    // NL=5 -> final hidden is hidA
    inr_last_kernel<<<B_SZ / 8, 256, 0, stream>>>(hidA, lastW, lastB, (float*)d_out);
}
